// CovAndHW_16415365006036
// MI455X (gfx1250) — compile-verified, matrix operand order checked by reading, not run on hardware
//
#include <hip/hip_runtime.h>
#include <hip/hip_bf16.h>

typedef __attribute__((ext_vector_type(16))) __bf16 v16bf;
typedef __attribute__((ext_vector_type(8)))  __bf16 v8bf;
typedef __attribute__((ext_vector_type(8)))  float  v8f;

#define SIZE 160
#define CCH  64
#define BATCH 8
#define KDIM (CCH * SIZE)   // 10240

// ---------------------------------------------------------------------------
// Kernel 1: nearest resize 256->160 + bf16 pack into A [h][(c,w)] and
// At [w][(c,h)] (transposed through LDS so both writes are coalesced).
// ---------------------------------------------------------------------------
__global__ __launch_bounds__(256)
void pack_x_kernel(const float* __restrict__ x,
                   __bf16* __restrict__ Abuf,
                   __bf16* __restrict__ Atbuf) {
    const int bc    = blockIdx.x;          // 0..511
    const int batch = bc >> 6;
    const int ch    = bc & 63;
    const float* xs = x + (size_t)bc * 256 * 256;

    __shared__ __bf16 tile[SIZE][SIZE + 1];   // 51.5 KB

    for (int it = 0; it < 100; ++it) {
        int idx = it * 256 + threadIdx.x;     // 0..25599
        int h = idx / SIZE, w = idx - h * SIZE;
        int hi = (h * 8) / 5;                 // floor(h*256/160)
        int wi = (w * 8) / 5;
        __bf16 v = (__bf16)xs[hi * 256 + wi];
        Abuf[((size_t)batch * SIZE + h) * KDIM + ch * SIZE + w] = v;
        tile[h][w] = v;
    }
    __syncthreads();
    for (int it = 0; it < 100; ++it) {
        int idx = it * 256 + threadIdx.x;
        int w = idx / SIZE, h = idx - w * SIZE;
        Atbuf[((size_t)batch * SIZE + w) * KDIM + ch * SIZE + h] = tile[h][w];
    }
}

// ---------------------------------------------------------------------------
// Kernel 2: weights fp32 -> bf16 (layouts already [o][(c,k)] row-major).
// ---------------------------------------------------------------------------
__global__ __launch_bounds__(256)
void pack_w_kernel(const float* __restrict__ Wb, const float* __restrict__ Wc,
                   __bf16* __restrict__ Wb16, __bf16* __restrict__ Wc16) {
    int i = blockIdx.x * 256 + threadIdx.x;
    const int N = CCH * KDIM;                 // 655360
    if (i < N) {
        Wb16[i] = (__bf16)Wb[i];
        Wc16[i] = (__bf16)Wc[i];
    }
}

// ---------------------------------------------------------------------------
// Kernel 3: the two GEMMs (b and c), M=160 N=64 K=10240 per batch, via
// v_wmma_f32_16x16x32_bf16 with a software-pipelined K loop. Each wave owns
// TWO 16x16 M-tiles sharing one B fragment (2 WMMAs per load group), fused
// folded-BN scale/bias + SiLU epilogue.
// grid = 2 gemms * 8 batches * 5 mtile-pairs = 80 blocks, 4 waves / block.
// ---------------------------------------------------------------------------
__global__ __launch_bounds__(128)
void gemm_silu_kernel(const __bf16* __restrict__ Abuf,
                      const __bf16* __restrict__ Atbuf,
                      const __bf16* __restrict__ Wb16,
                      const __bf16* __restrict__ Wc16,
                      const float* __restrict__ gb, const float* __restrict__ bb,
                      const float* __restrict__ gc, const float* __restrict__ bcv,
                      float* __restrict__ bvec, float* __restrict__ cvec) {
    int bid   = blockIdx.x;
    int g     = bid / 40;                 // 0 = "b" gemm, 1 = "c" gemm
    int rem   = bid - g * 40;
    int batch = rem / 5;
    int mtp   = rem - batch * 5;          // m-tile pair index: rows [32*mtp, 32*mtp+32)

    int wave  = threadIdx.x >> 5;
    int lane  = threadIdx.x & 31;
    int lhalf = lane & 15;
    int m0a   = mtp * 32;
    int m0b   = m0a + 16;
    int n0    = wave * 16;
    int off   = (lane < 16) ? 0 : 8;      // K sub-offset per A/B bf16 layout

    const __bf16* A    = (g == 0 ? Abuf : Atbuf) + (size_t)batch * SIZE * KDIM;
    const __bf16* W    = (g == 0 ? Wb16 : Wc16);
    const float*  gain = (g == 0 ? gb : gc);
    const float*  bias = (g == 0 ? bb : bcv);
    float*        outv = (g == 0 ? bvec : cvec) + (size_t)batch * CCH * SIZE;

    // A fragment: lanes 0-15 & 16-31 both hold rows (lhalf), with K
    // {off..off+7} and {off+16..off+23}: two contiguous 16B loads per tile.
    // B fragment (W columns contiguous in K): same addressing, col n0+lhalf.
    const __bf16* ArowA = A + (size_t)(m0a + lhalf) * KDIM + off;
    const __bf16* ArowB = A + (size_t)(m0b + lhalf) * KDIM + off;
    const __bf16* Wrow  = W + (size_t)(n0  + lhalf) * KDIM + off;

    v8f accA = {};
    v8f accB = {};

    // -------- software-pipelined K loop (single-stage double buffer) -------
    v8bf a0A = *(const v8bf*)(ArowA);
    v8bf a1A = *(const v8bf*)(ArowA + 16);
    v8bf a0B = *(const v8bf*)(ArowB);
    v8bf a1B = *(const v8bf*)(ArowB + 16);
    v8bf b0  = *(const v8bf*)(Wrow);
    v8bf b1  = *(const v8bf*)(Wrow + 16);

    for (int kb = 0; kb < KDIM - 32; kb += 32) {
        v16bf amatA = __builtin_shufflevector(a0A, a1A, 0,1,2,3,4,5,6,7,8,9,10,11,12,13,14,15);
        v16bf amatB = __builtin_shufflevector(a0B, a1B, 0,1,2,3,4,5,6,7,8,9,10,11,12,13,14,15);
        v16bf bmat  = __builtin_shufflevector(b0,  b1,  0,1,2,3,4,5,6,7,8,9,10,11,12,13,14,15);
        // branch-free near-cache streaming prefetch ~1KB ahead (stays in d_ws)
        __builtin_prefetch(ArowA + kb + 544, 0, 3);
        __builtin_prefetch(ArowB + kb + 544, 0, 3);
        __builtin_prefetch(Wrow  + kb + 544, 0, 3);
        // loads for kb+32 overlap the WMMAs for kb
        a0A = *(const v8bf*)(ArowA + kb + 32);
        a1A = *(const v8bf*)(ArowA + kb + 48);
        a0B = *(const v8bf*)(ArowB + kb + 32);
        a1B = *(const v8bf*)(ArowB + kb + 48);
        b0  = *(const v8bf*)(Wrow  + kb + 32);
        b1  = *(const v8bf*)(Wrow  + kb + 48);
        accA = __builtin_amdgcn_wmma_f32_16x16x32_bf16(
                   false, amatA, false, bmat, (short)0, accA, false, false);
        accB = __builtin_amdgcn_wmma_f32_16x16x32_bf16(
                   false, amatB, false, bmat, (short)0, accB, false, false);
    }
    {   // drain
        v16bf amatA = __builtin_shufflevector(a0A, a1A, 0,1,2,3,4,5,6,7,8,9,10,11,12,13,14,15);
        v16bf amatB = __builtin_shufflevector(a0B, a1B, 0,1,2,3,4,5,6,7,8,9,10,11,12,13,14,15);
        v16bf bmat  = __builtin_shufflevector(b0,  b1,  0,1,2,3,4,5,6,7,8,9,10,11,12,13,14,15);
        accA = __builtin_amdgcn_wmma_f32_16x16x32_bf16(
                   false, amatA, false, bmat, (short)0, accA, false, false);
        accB = __builtin_amdgcn_wmma_f32_16x16x32_bf16(
                   false, amatB, false, bmat, (short)0, accB, false, false);
    }

    // C/D layout: VGPR r, lanes 0-15 -> (M=m0+r,   N=n0+lane),
    //                     lanes 16-31 -> (M=m0+8+r, N=n0+lane-16)
    int   o      = n0 + lhalf;
    float gn     = gain[o];
    float bs     = bias[o];
    int   hiHalf = (lane < 16) ? 0 : 8;
    int   mbaseA = m0a + hiHalf;
    int   mbaseB = m0b + hiHalf;
#pragma unroll
    for (int r = 0; r < 8; ++r) {
        float yA = accA[r] * gn + bs;
        float yB = accB[r] * gn + bs;
        outv[(size_t)o * SIZE + (mbaseA + r)] = yA / (1.0f + __expf(-yA)); // SiLU
        outv[(size_t)o * SIZE + (mbaseB + r)] = yB / (1.0f + __expf(-yB));
    }
}

// ---------------------------------------------------------------------------
// Kernel 4: per (batch,channel): s = <b,c>; t = X @ b;  out256 upsample of
// X + s * t (x) c.   X staged in dynamic LDS (102.4 KB).
// ---------------------------------------------------------------------------
__global__ __launch_bounds__(256)
void finalize_kernel(const float* __restrict__ x,
                     const float* __restrict__ bvec,
                     const float* __restrict__ cvec,
                     float* __restrict__ out) {
    extern __shared__ float sm[];
    float* Xs  = sm;                 // 25600
    float* bv  = Xs + SIZE * SIZE;   // 160
    float* cv  = bv + SIZE;          // 160
    float* tv  = cv + SIZE;          // 160
    float* red = tv + SIZE;          // 256

    const int bc  = blockIdx.x;      // batch*64 + channel
    const int tid = threadIdx.x;
    const float* xs  = x    + (size_t)bc * 256 * 256;
    const float* bvp = bvec + (size_t)bc * SIZE;
    const float* cvp = cvec + (size_t)bc * SIZE;

    if (tid < SIZE) { bv[tid] = bvp[tid]; cv[tid] = cvp[tid]; }
    for (int it = 0; it < 100; ++it) {
        int idx = it * 256 + tid;
        int h = idx / SIZE, w = idx - h * SIZE;
        Xs[idx] = xs[((h * 8) / 5) * 256 + ((w * 8) / 5)];
    }
    __syncthreads();

    // s = dot(b, c)
    red[tid] = (tid < SIZE) ? bv[tid] * cv[tid] : 0.0f;
    __syncthreads();
    for (int st = 128; st > 0; st >>= 1) {
        if (tid < st) red[tid] += red[tid + st];
        __syncthreads();
    }
    float s = red[0];

    // t[h] = sum_w X[h,w] * b[w]
    if (tid < SIZE) {
        const float* row = Xs + tid * SIZE;
        float acc = 0.0f;
        for (int w = 0; w < SIZE; ++w) acc += row[w] * bv[w];
        tv[tid] = acc;
    }
    __syncthreads();

    // upsample 160->256 with the fused rank-1 update
    float* op = out + (size_t)bc * 256 * 256;
    int ws = (tid * 5) >> 3;                 // floor(col*160/256)
    float cw = cv[ws];
    for (int r = 0; r < 256; ++r) {
        int hs = (r * 5) >> 3;
        op[r * 256 + tid] = Xs[hs * SIZE + ws] + s * tv[hs] * cw;
    }
}

// ---------------------------------------------------------------------------
extern "C" void kernel_launch(void* const* d_in, const int* in_sizes, int n_in,
                              void* d_out, int out_size, void* d_ws, size_t ws_size,
                              hipStream_t stream) {
    const float* x   = (const float*)d_in[0];
    const float* Wb  = (const float*)d_in[1];
    const float* Wc  = (const float*)d_in[2];
    const float* gb  = (const float*)d_in[3];
    const float* bb  = (const float*)d_in[4];
    const float* gc  = (const float*)d_in[5];
    const float* bcv = (const float*)d_in[6];
    float* out = (float*)d_out;

    // workspace carve-up
    const size_t SA = (size_t)BATCH * SIZE * KDIM;   // 13,107,200 bf16 elems
    const size_t SW = (size_t)CCH * KDIM;            //    655,360 bf16 elems
    const size_t SV = (size_t)BATCH * CCH * SIZE;    //     81,920 f32 elems
    char* p = (char*)d_ws;
    __bf16* Abuf  = (__bf16*)p;  p += SA * 2;
    __bf16* Atbuf = (__bf16*)p;  p += SA * 2;
    __bf16* Wb16  = (__bf16*)p;  p += SW * 2;
    __bf16* Wc16  = (__bf16*)p;  p += SW * 2;
    float*  bvec  = (float*)p;   p += SV * 4;
    float*  cvec  = (float*)p;   p += SV * 4;       // total ~53.1 MiB

    pack_x_kernel<<<BATCH * CCH, 256, 0, stream>>>(x, Abuf, Atbuf);
    pack_w_kernel<<<(CCH * KDIM + 255) / 256, 256, 0, stream>>>(Wb, Wc, Wb16, Wc16);
    gemm_silu_kernel<<<2 * BATCH * 5, 128, 0, stream>>>(
        Abuf, Atbuf, Wb16, Wc16, gb, bb, gc, bcv, bvec, cvec);
    const size_t smem = (size_t)(SIZE * SIZE + 3 * SIZE + 256) * sizeof(float);
    finalize_kernel<<<BATCH * CCH, 256, smem, stream>>>(x, bvec, cvec, out);
}
